// ConvBNBlock_60902636257545
// MI455X (gfx1250) — matrix operand order repaired and checked by reading, not compiled, not run on hardware
//
#include <hip/hip_runtime.h>
#include <hip/hip_bf16.h>

typedef __attribute__((ext_vector_type(16))) __bf16 v16bf;
typedef __attribute__((ext_vector_type(8)))  float  v8f;

#define C_IN  64
#define C_OUT 64
#define NKOFF 27
#define BN_EPS 1e-4f
#define TPW 4   // tiles (of 16 rulebook rows) processed per wave, reusing B fragments

__device__ __forceinline__ unsigned short f32_to_bf16_bits(float x) {
    union { float f; unsigned u; } t; t.f = x;
    unsigned r = t.u + 0x7FFFu + ((t.u >> 16) & 1u);   // round-to-nearest-even
    return (unsigned short)(r >> 16);
}

// ---------------------------------------------------------------------------
// Phase 0: zero accumulator, convert features f32 -> bf16 (row N zeroed),
// zero BN sums.
// ---------------------------------------------------------------------------
__global__ __launch_bounds__(256) void prep_kernel(
    const float* __restrict__ feat,      // N x 64
    unsigned short* __restrict__ fbf,    // (N+1) x 64  (bf16 bits)
    float* __restrict__ acc,             // (N+1) x 64
    float* __restrict__ bnsum,           // 128
    long total, long nfeat)
{
    long i = (long)blockIdx.x * 256 + threadIdx.x;
    if (i < total) {
        acc[i] = 0.0f;
        fbf[i] = (i < nfeat) ? f32_to_bf16_bits(feat[i]) : (unsigned short)0;
    }
    if (blockIdx.x == 0 && threadIdx.x < 128) bnsum[threadIdx.x] = 0.0f;
}

// ---------------------------------------------------------------------------
// Phase 1: pack W (27 x Cin x Cout, f32) into WMMA B-fragment layout (bf16).
// Fragment f = c*4 + n  (c = K-chunk of 32, n = N-tile of 16).
// B 32x16 packing used here (contiguous k, not AMD's table order): lane l holds column n*16 + (l&15),
//   K-half (l>>4)*16, 16 contiguous K values per lane.
// One thread per (k, f, lane): 27*8*32 = 6912 threads.
// ---------------------------------------------------------------------------
__global__ __launch_bounds__(256) void packW_kernel(
    const float* __restrict__ W,              // 27 x 64 x 64
    unsigned short* __restrict__ Wpack)       // 27 x 8 x 32 x 16 (bf16 bits)
{
    int idx = blockIdx.x * 256 + threadIdx.x;     // = k*256 + f*32 + lane
    if (idx >= NKOFF * 8 * 32) return;
    int lane = idx & 31;
    int f    = (idx >> 5) & 7;
    int k    = idx >> 8;
    int c = f >> 2, n = f & 3;
    int khalf = (lane >> 4) * 16;
    int col   = n * 16 + (lane & 15);
    unsigned short* dst = Wpack + (long)idx * 16;
#pragma unroll
    for (int j = 0; j < 16; ++j) {
        int cin = c * 32 + khalf + j;
        dst[j] = f32_to_bf16_bits(W[((long)k * C_IN + cin) * C_OUT + col]);
    }
}

// ---------------------------------------------------------------------------
// Phase 2: gather -> WMMA bf16 GEMM -> scatter-add.
// One wave handles TPW consecutive 16-row tiles of one kernel offset k.
// ---------------------------------------------------------------------------
__global__ __launch_bounds__(256) void conv_wmma_kernel(
    const unsigned short* __restrict__ feats,  // (N+1) x 64 bf16 bits
    const unsigned short* __restrict__ Wpack,  // 27 x 8 x 32 x 16 bf16 bits
    const int* __restrict__ in_idx,            // 27 x P
    const int* __restrict__ out_idx,           // 27 x P
    float* __restrict__ acc,                   // (N+1) x 64
    int N, int P, int chunksPerK, int totalChunks)
{
    int wave = (blockIdx.x * 256 + threadIdx.x) >> 5;   // wave-uniform
    if (wave >= totalChunks) return;
    int lane = threadIdx.x & 31;
    int k     = wave / chunksPerK;
    int chunk = wave - k * chunksPerK;
    int tile0 = chunk * TPW;
    int tilesPerK = P >> 4;                    // P is a multiple of 16

    // Padding fills the tail of each k-row: if the first row of the first tile
    // is padding, every row this wave would touch is padding. Uniform branch.
    if (in_idx[(long)k * P + (long)tile0 * 16] == N) return;

    int mlo = lane & 15;     // A gather row within tile; also N column for B/C/D
    int hi  = lane >> 4;     // lane half

    // Load the 8 B fragments for this k (reused across TPW tiles).
    v16bf b[8];
    {
        const unsigned short* wp = Wpack + (long)k * 8 * 32 * 16;
#pragma unroll
        for (int f = 0; f < 8; ++f) {
            union { uint4 u[2]; v16bf v; } t;
            const uint4* p = (const uint4*)(wp + ((long)(f * 32 + lane)) * 16);
            t.u[0] = p[0];
            t.u[1] = p[1];
            b[f] = t.v;
        }
    }

#pragma unroll 1
    for (int t = 0; t < TPW; ++t) {
        int tile = tile0 + t;
        if (tile >= tilesPerK) break;                       // uniform
        long base = (long)k * P + (long)tile * 16;
        if (t > 0 && in_idx[base] == N) break;              // uniform: tail padding

        // ---- gather A (16 x 64 bf16) as two 16x32 fragments ----
        // A 16x32 layout: lane l, row M=l&15; lane half selects K-offset 0 or 16;
        // regs 0-7 hold K = off..off+15 (contiguous, matching packed B).
        int arow = in_idx[base + mlo];                      // padded rows -> N (zero row)
        const unsigned short* aptr = feats + (long)arow * C_IN;
        v16bf a[2];
#pragma unroll
        for (int c = 0; c < 2; ++c) {
            union { uint4 u[2]; v16bf v; } ta;
            ta.u[0] = *(const uint4*)(aptr + c * 32 + hi * 16);
            ta.u[1] = *(const uint4*)(aptr + c * 32 + hi * 16 + 8);
            a[c] = ta.v;
        }

        // ---- 8x v_wmma_f32_16x16x32_bf16 ----
        v8f cf[4] = {};
#pragma unroll
        for (int n = 0; n < 4; ++n) {
            cf[n] = __builtin_amdgcn_wmma_f32_16x16x32_bf16(
                        false, a[0], false, b[0 * 4 + n], (short)0, cf[n], false, false);
            cf[n] = __builtin_amdgcn_wmma_f32_16x16x32_bf16(
                        false, a[1], false, b[1 * 4 + n], (short)0, cf[n], false, false);
        }

        // ---- scatter-add (C/D layout: lane l, reg r -> M=(l>>4)*8+r, N=l&15) ----
        int orow[8];
#pragma unroll
        for (int r = 0; r < 8; ++r)
            orow[r] = out_idx[base + hi * 8 + r];           // padding -> dump row N

#pragma unroll
        for (int n = 0; n < 4; ++n) {
            int col = n * 16 + mlo;
#pragma unroll
            for (int r = 0; r < 8; ++r)
                atomicAdd(acc + (long)orow[r] * C_OUT + col, cf[n][r]);
        }
    }
}

// ---------------------------------------------------------------------------
// Phase 3: per-channel sum / sum-of-squares over N rows.
// ---------------------------------------------------------------------------
__global__ __launch_bounds__(256) void bn_reduce_kernel(
    const float* __restrict__ acc, float* __restrict__ bnsum, int N)
{
    __shared__ float ls[256], ls2[256];
    int c = threadIdx.x & 63;
    int g = threadIdx.x >> 6;   // 0..3
    float s = 0.0f, s2 = 0.0f;
    for (long row = (long)blockIdx.x * 4 + g; row < N; row += (long)gridDim.x * 4) {
        float v = acc[row * C_OUT + c];
        s += v; s2 += v * v;
    }
    ls[threadIdx.x] = s; ls2[threadIdx.x] = s2;
    __syncthreads();
    if (g == 0) {
        s  = ls[c]  + ls[64 + c]  + ls[128 + c]  + ls[192 + c];
        s2 = ls2[c] + ls2[64 + c] + ls2[128 + c] + ls2[192 + c];
        atomicAdd(bnsum + c, s);
        atomicAdd(bnsum + 64 + c, s2);
    }
}

// ---------------------------------------------------------------------------
// Phase 4: BN normalize + scale/shift + ReLU.
// ---------------------------------------------------------------------------
__global__ __launch_bounds__(256) void bn_apply_kernel(
    const float* __restrict__ acc, const float* __restrict__ bnsum,
    const float* __restrict__ gamma, const float* __restrict__ beta,
    float* __restrict__ out, long nfeat, float invN)
{
    long i = (long)blockIdx.x * 256 + threadIdx.x;
    if (i >= nfeat) return;
    int c = (int)(i & 63);
    float mean = bnsum[c] * invN;
    float var  = bnsum[64 + c] * invN - mean * mean;
    float rs   = rsqrtf(var + BN_EPS);
    float y    = (acc[i] - mean) * rs * gamma[c] + beta[c];
    out[i] = fmaxf(y, 0.0f);
}

// ---------------------------------------------------------------------------
extern "C" void kernel_launch(void* const* d_in, const int* in_sizes, int n_in,
                              void* d_out, int out_size, void* d_ws, size_t ws_size,
                              hipStream_t stream)
{
    const float* features = (const float*)d_in[0];   // N x 64
    const float* W        = (const float*)d_in[1];   // 27 x 64 x 64
    const float* gamma    = (const float*)d_in[2];   // 64
    const float* beta     = (const float*)d_in[3];   // 64
    const int*   in_idx   = (const int*)d_in[4];     // 27 x P
    const int*   out_idx  = (const int*)d_in[5];     // 27 x P
    float*       out      = (float*)d_out;

    const int N = in_sizes[0] / C_IN;
    const int P = in_sizes[4] / NKOFF;

    // Workspace carve-up (256-byte aligned).
    char* ws = (char*)d_ws;
    size_t off = 0;
    auto carve = [&](size_t bytes) { size_t o = off; off = (off + bytes + 255) & ~(size_t)255; return o; };
    const long totalElems = (long)(N + 1) * C_IN;          // padded with dump/zero row
    float*          acc   = (float*)(ws + carve(totalElems * sizeof(float)));
    unsigned short* fbf   = (unsigned short*)(ws + carve(totalElems * sizeof(unsigned short)));
    unsigned short* wpack = (unsigned short*)(ws + carve((size_t)NKOFF * 8 * 32 * 16 * sizeof(unsigned short)));
    float*          bnsum = (float*)(ws + carve(128 * sizeof(float)));

    // Phase 0: zero acc, features -> bf16, zero BN sums.
    {
        long blocks = (totalElems + 255) / 256;
        prep_kernel<<<(int)blocks, 256, 0, stream>>>(features, fbf, acc, bnsum,
                                                     totalElems, (long)N * C_IN);
    }
    // Phase 1: pack W into WMMA B-fragment layout (27 blocks * 256 = 6912 threads).
    packW_kernel<<<NKOFF, 256, 0, stream>>>(W, wpack);

    // Phase 2: gather-GEMM-scatter with bf16 WMMA.
    {
        int tilesPerK   = P >> 4;                            // P % 16 == 0 for this problem
        int chunksPerK  = (tilesPerK + TPW - 1) / TPW;
        int totalChunks = NKOFF * chunksPerK;
        int blocks      = (totalChunks + 7) / 8;             // 8 waves per 256-thread block
        conv_wmma_kernel<<<blocks, 256, 0, stream>>>(fbf, wpack, in_idx, out_idx,
                                                     acc, N, P, chunksPerK, totalChunks);
    }

    // Phase 3: BN statistics.
    bn_reduce_kernel<<<1024, 256, 0, stream>>>(acc, bnsum, N);

    // Phase 4: normalize + ReLU into d_out.
    {
        long nfeat  = (long)N * C_OUT;
        long blocks = (nfeat + 255) / 256;
        bn_apply_kernel<<<(int)blocks, 256, 0, stream>>>(acc, bnsum, gamma, beta,
                                                         out, nfeat, 1.0f / (float)N);
    }
}
